// AutodiffCHAREE_69114613727867
// MI455X (gfx1250) — compile-verified, matrix operand order checked by reading, not run on hardware
//
#include <hip/hip_runtime.h>
#include <hip/hip_bf16.h>

// ---------------------------------------------------------------------------
// Problem constants
// ---------------------------------------------------------------------------
#define DD       1024        // density matrix dimension (32*32)
#define KSTATES  2048        // number of product states
#define BM       128         // block tile M (8 waves x 16)
#define BN       32          // block tile N (2 sub-tiles of 16 per wave)
#define BK       32          // K step (WMMA f16 K)
#define LDK      40          // padded LDS row stride in halves (80B: 16B-aligned rows)

typedef __attribute__((ext_vector_type(16))) _Float16     v16h;
typedef __attribute__((ext_vector_type(8)))  float        v8f;
typedef __attribute__((ext_vector_type(8)))  unsigned int v8u;

union FragU { v8u u; v16h h; };
struct alignas(16) H8 { _Float16 h[8]; };

// Gather a 16x32 f16 WMMA operand fragment from an LDS tile stored [row][k]
// with row stride LDK halves.  Per CDNA5 ISA 7.12.2 (16-bit A-matrix 16x32):
//   lane 0-15: M=lane, K base 0 ; lane 16-31: M=lane-16, K base 8
//   VGPR v in 0..3 holds K = {base+2v, base+2v+1}; v in 4..7 same +16.
__device__ __forceinline__ v16h load_frag(const _Float16* tile, int row0, int lane) {
    const unsigned int* pr =
        (const unsigned int*)(tile + (size_t)(row0 + (lane & 15)) * LDK);
    const int kb2 = (lane >> 4) << 2;   // dword offset: 0 or 4
    FragU f;
#pragma unroll
    for (int v = 0; v < 4; ++v) f.u[v]     = pr[kb2 + v];
#pragma unroll
    for (int v = 0; v < 4; ++v) f.u[4 + v] = pr[8 + kb2 + v];
    return f.h;
}

#define WMMA_F16(A, B, C) \
    __builtin_amdgcn_wmma_f32_16x16x32_f16(false, (A), false, (B), (short)0, (C), false, false)

// ---------------------------------------------------------------------------
// Complex GEMM (f32 planes, f16 WMMA compute):  C = A * B,  all DD x DD
// Block: 256 thr = 8 waves; block tile 128x32; wave tile 16x32 (2 N sub-tiles,
// A fragments reused across both -> 8 WMMA per 12 ds_load_b128 per wave/step).
// ---------------------------------------------------------------------------
__global__ void __launch_bounds__(256)
cgemm_nn_kernel(float* __restrict__ Cre, float* __restrict__ Cim,
                const float* __restrict__ Are, const float* __restrict__ Aim,
                const float* __restrict__ Bre, const float* __restrict__ Bim)
{
    __shared__ _Float16 lAr[BM * LDK], lAi[BM * LDK];   // 10240B each
    __shared__ _Float16 lBr[BN * LDK], lBi[BN * LDK];   // 2560B each

    const int bm   = blockIdx.x * BM;
    const int bn   = blockIdx.y * BN;
    const int tid  = threadIdx.x;
    const int lane = tid & 31;
    const int wr   = tid >> 5;    // wave id = M sub-tile 0..7

    v8f p1[2] = {v8f{}, v8f{}}, p2[2] = {v8f{}, v8f{}};
    v8f p3[2] = {v8f{}, v8f{}}, p4[2] = {v8f{}, v8f{}};

    for (int k0 = 0; k0 < DD; k0 += BK) {
        __syncthreads();
        {   // stage A[m][k] -> f16, float4 (16B) global loads along k
            const int kq    = (tid & 7) * 4;   // 0,4,...,28
            const int rbase = tid >> 3;        // 0..31
#pragma unroll
            for (int q = 0; q < 4; ++q) {
                const int m = q * 32 + rbase;
                const float4 vr = *(const float4*)&Are[(size_t)(bm + m) * DD + k0 + kq];
                const float4 vi = *(const float4*)&Aim[(size_t)(bm + m) * DD + k0 + kq];
                _Float16* dr = &lAr[m * LDK + kq];
                _Float16* di = &lAi[m * LDK + kq];
                dr[0] = (_Float16)vr.x; dr[1] = (_Float16)vr.y;
                dr[2] = (_Float16)vr.z; dr[3] = (_Float16)vr.w;
                di[0] = (_Float16)vi.x; di[1] = (_Float16)vi.y;
                di[2] = (_Float16)vi.z; di[3] = (_Float16)vi.w;
            }
            // stage B transposed: Bt[n][k] = B[k][n]; float4 loads along n
            const int kk = tid >> 3;           // 0..31
            const int n4 = (tid & 7) * 4;      // 0,4,...,28
            const float4 br4 = *(const float4*)&Bre[(size_t)(k0 + kk) * DD + bn + n4];
            const float4 bi4 = *(const float4*)&Bim[(size_t)(k0 + kk) * DD + bn + n4];
            lBr[(n4 + 0) * LDK + kk] = (_Float16)br4.x;
            lBr[(n4 + 1) * LDK + kk] = (_Float16)br4.y;
            lBr[(n4 + 2) * LDK + kk] = (_Float16)br4.z;
            lBr[(n4 + 3) * LDK + kk] = (_Float16)br4.w;
            lBi[(n4 + 0) * LDK + kk] = (_Float16)bi4.x;
            lBi[(n4 + 1) * LDK + kk] = (_Float16)bi4.y;
            lBi[(n4 + 2) * LDK + kk] = (_Float16)bi4.z;
            lBi[(n4 + 3) * LDK + kk] = (_Float16)bi4.w;
            if (k0 + BK < DD)   // pull next A tile toward L2/L0
                __builtin_prefetch(&Are[(size_t)(bm + rbase) * DD + k0 + BK], 0, 1);
        }
        __syncthreads();

        const v16h aR = load_frag(lAr, wr * 16, lane);
        const v16h aI = load_frag(lAi, wr * 16, lane);
#pragma unroll
        for (int s = 0; s < 2; ++s) {
            const v16h bR = load_frag(lBr, s * 16, lane);
            const v16h bI = load_frag(lBi, s * 16, lane);
            p1[s] = WMMA_F16(aR, bR, p1[s]);
            p2[s] = WMMA_F16(aI, bI, p2[s]);
            p3[s] = WMMA_F16(aR, bI, p3[s]);
            p4[s] = WMMA_F16(aI, bR, p4[s]);
        }
    }

    // C/D layout: lane n = lane&15, row = r + 8*(lane>=16)
    const int mb = bm + wr * 16 + ((lane >> 4) << 3);
#pragma unroll
    for (int s = 0; s < 2; ++s) {
        const int n = bn + s * 16 + (lane & 15);
#pragma unroll
        for (int r = 0; r < 8; ++r) {
            Cre[(size_t)(mb + r) * DD + n] = p1[s][r] - p2[s][r];   // Re(AB)
            Cim[(size_t)(mb + r) * DD + n] = p3[s][r] + p4[s][r];   // Im(AB)
        }
    }
}

// ---------------------------------------------------------------------------
// rho = Psi^T conj(Psi):  rho[m][n] = sum_k Psi[k][m] * conj(Psi[k][n])
// Psi stored as f16 planes (KSTATES x DD); A is a transposed-source tile.
// ---------------------------------------------------------------------------
__global__ void __launch_bounds__(256)
rho_gemm_kernel(float* __restrict__ Cre, float* __restrict__ Cim,
                const _Float16* __restrict__ PsiR, const _Float16* __restrict__ PsiI)
{
    __shared__ _Float16 lAr[BM * LDK], lAi[BM * LDK];
    __shared__ _Float16 lBr[BN * LDK], lBi[BN * LDK];

    const int bm   = blockIdx.x * BM;
    const int bn   = blockIdx.y * BN;
    const int tid  = threadIdx.x;
    const int lane = tid & 31;
    const int wr   = tid >> 5;

    v8f p1[2] = {v8f{}, v8f{}}, p2[2] = {v8f{}, v8f{}};
    v8f p3[2] = {v8f{}, v8f{}}, p4[2] = {v8f{}, v8f{}};

    for (int k0 = 0; k0 < KSTATES; k0 += BK) {
        __syncthreads();
        {   // A tile (transposed source): lA[m][kk] = Psi[k0+kk][bm+m]
            // 16B loads of 8 halves along m, scattered f16 stores to LDS rows
            const int mm8 = (tid & 15) * 8;    // 0,8,...,120
            const int kkb = tid >> 4;          // 0..15
#pragma unroll
            for (int q = 0; q < 2; ++q) {
                const int kk = q * 16 + kkb;
                const H8 hr = *(const H8*)&PsiR[(size_t)(k0 + kk) * DD + bm + mm8];
                const H8 hi = *(const H8*)&PsiI[(size_t)(k0 + kk) * DD + bm + mm8];
#pragma unroll
                for (int i = 0; i < 8; ++i) {
                    lAr[(mm8 + i) * LDK + kk] = hr.h[i];
                    lAi[(mm8 + i) * LDK + kk] = hi.h[i];
                }
            }
            if (tid < 128) {   // B tile: lB[n][kk] = Psi[k0+kk][bn+n]
                const int kk = tid >> 2;       // 0..31
                const int n8 = (tid & 3) * 8;  // 0,8,16,24
                const H8 hr = *(const H8*)&PsiR[(size_t)(k0 + kk) * DD + bn + n8];
                const H8 hi = *(const H8*)&PsiI[(size_t)(k0 + kk) * DD + bn + n8];
#pragma unroll
                for (int i = 0; i < 8; ++i) {
                    lBr[(n8 + i) * LDK + kk] = hr.h[i];
                    lBi[(n8 + i) * LDK + kk] = hi.h[i];
                }
            }
        }
        __syncthreads();

        const v16h aR = load_frag(lAr, wr * 16, lane);
        const v16h aI = load_frag(lAi, wr * 16, lane);
#pragma unroll
        for (int s = 0; s < 2; ++s) {
            const v16h bR = load_frag(lBr, s * 16, lane);
            const v16h bI = load_frag(lBi, s * 16, lane);
            p1[s] = WMMA_F16(aR, bR, p1[s]);
            p2[s] = WMMA_F16(aI, bI, p2[s]);
            p3[s] = WMMA_F16(aI, bR, p3[s]);
            p4[s] = WMMA_F16(aR, bI, p4[s]);
        }
    }

    const int mb = bm + wr * 16 + ((lane >> 4) << 3);
#pragma unroll
    for (int s = 0; s < 2; ++s) {
        const int n = bn + s * 16 + (lane & 15);
#pragma unroll
        for (int r = 0; r < 8; ++r) {
            // (R+iI)(R'-iI'): Re = RR'+II', Im = IR'-RI'
            Cre[(size_t)(mb + r) * DD + n] = p1[s][r] + p2[s][r];
            Cim[(size_t)(mb + r) * DD + n] = p3[s][r] - p4[s][r];
        }
    }
}

// ---------------------------------------------------------------------------
// Elementwise helpers over complex planes (exactly DD*DD threads launched)
// ---------------------------------------------------------------------------
__global__ void __launch_bounds__(256)
affine_diag_kernel(float* __restrict__ Wre, float* __restrict__ Wim,
                   const float* __restrict__ Tre, const float* __restrict__ Tim,
                   float alpha, float beta)   // W = alpha*T + beta*I
{
    const int idx = blockIdx.x * 256 + threadIdx.x;
    const int i = idx >> 10, j = idx & 1023;
    Wre[idx] = alpha * Tre[idx] + ((i == j) ? beta : 0.0f);
    Wim[idx] = alpha * Tim[idx];
}

__global__ void __launch_bounds__(256)
axpy_kernel(float* __restrict__ Yre, float* __restrict__ Yim,
            const float* __restrict__ Xre, const float* __restrict__ Xim,
            float a, float b)                 // Y = a*X + b*Y
{
    const int idx = blockIdx.x * 256 + threadIdx.x;
    Yre[idx] = a * Xre[idx] + b * Yre[idx];
    Yim[idx] = a * Xim[idx] + b * Yim[idx];
}

__global__ void __launch_bounds__(256)
ident_kernel(float* __restrict__ Rre, float* __restrict__ Rim)
{
    const int idx = blockIdx.x * 256 + threadIdx.x;
    const int i = idx >> 10, j = idx & 1023;
    Rre[idx] = (i == j) ? 1.0f : 0.0f;
    Rim[idx] = 0.0f;
}

__global__ void __launch_bounds__(256)
conv_rt_kernel(float* __restrict__ Rre, float* __restrict__ Rim,
               const double* __restrict__ rt)   // complex128 interleaved -> planes
{
    const int idx = blockIdx.x * 256 + threadIdx.x;
    Rre[idx] = (float)rt[2 * (size_t)idx];
    Rim[idx] = (float)rt[2 * (size_t)idx + 1];
}

// ---------------------------------------------------------------------------
// softmax over theta_p (single block)
// ---------------------------------------------------------------------------
__global__ void __launch_bounds__(256)
softmax_kernel(float* __restrict__ p, const float* __restrict__ x)
{
    __shared__ float red[256];
    const int tid = threadIdx.x;
    float mx = -1e30f;
    for (int i = tid; i < KSTATES; i += 256) mx = fmaxf(mx, x[i]);
    red[tid] = mx; __syncthreads();
    for (int s = 128; s > 0; s >>= 1) { if (tid < s) red[tid] = fmaxf(red[tid], red[tid + s]); __syncthreads(); }
    mx = red[0]; __syncthreads();
    float sum = 0.0f;
    for (int i = tid; i < KSTATES; i += 256) sum += expf(x[i] - mx);
    red[tid] = sum; __syncthreads();
    for (int s = 128; s > 0; s >>= 1) { if (tid < s) red[tid] += red[tid + s]; __syncthreads(); }
    const float inv = 1.0f / red[0];
    for (int i = tid; i < KSTATES; i += 256) p[i] = expf(x[i] - mx) * inv;
}

// ---------------------------------------------------------------------------
// Build Psi[k][:] = sqrt(p_k) * kron(a_k, b_k) with a,b normalized (f16 planes)
// ---------------------------------------------------------------------------
__global__ void __launch_bounds__(128)
build_psi_kernel(_Float16* __restrict__ PsiR, _Float16* __restrict__ PsiI,
                 const float* __restrict__ thA, const float* __restrict__ thB,
                 const float* __restrict__ p)
{
    __shared__ float ar[32], ai[32], br[32], bi[32];
    __shared__ float red[64];
    const int k = blockIdx.x, tid = threadIdx.x;
    if (tid < 32)       { ar[tid] = thA[k * 64 + tid]; ai[tid] = thA[k * 64 + 32 + tid]; }
    else if (tid < 64)  { const int j = tid - 32; br[j] = thB[k * 64 + j]; bi[j] = thB[k * 64 + 32 + j]; }
    __syncthreads();
    if (tid < 32)       red[tid] = ar[tid] * ar[tid] + ai[tid] * ai[tid];
    else if (tid < 64)  { const int j = tid - 32; red[tid] = br[j] * br[j] + bi[j] * bi[j]; }
    __syncthreads();
    for (int s = 16; s > 0; s >>= 1) {
        if (tid < s) red[tid] += red[tid + s];
        else if (tid >= 32 && tid < 32 + s) red[tid] += red[tid + s];
        __syncthreads();
    }
    const float scale = sqrtf(p[k]) * rsqrtf(red[0]) * rsqrtf(red[32]);
    for (int t = tid; t < DD; t += 128) {
        const int i = t >> 5, j = t & 31;
        const float re = ar[i] * br[j] - ai[i] * bi[j];
        const float im = ar[i] * bi[j] + ai[i] * br[j];
        PsiR[(size_t)k * DD + t] = (_Float16)(scale * re);
        PsiI[(size_t)k * DD + t] = (_Float16)(scale * im);
    }
}

// ---------------------------------------------------------------------------
// t = Re tr(rho_t @ L)  — deterministic 2-stage reduction in f64
// ---------------------------------------------------------------------------
__global__ void __launch_bounds__(256)
trace_partial_kernel(double* __restrict__ part, const double* __restrict__ rt,
                     const float* __restrict__ Lre, const float* __restrict__ Lim)
{
    __shared__ double red[256];
    const int tid = threadIdx.x;
    double s = 0.0;
    const size_t total = (size_t)DD * DD;
    for (size_t idx = (size_t)blockIdx.x * 256 + tid; idx < total; idx += (size_t)gridDim.x * 256) {
        const int i = (int)(idx >> 10), j = (int)(idx & 1023);
        const size_t tji = (size_t)j * DD + i;
        s = fma(rt[2 * idx],      (double)Lre[tji], s);
        s = fma(-rt[2 * idx + 1], (double)Lim[tji], s);
    }
    red[tid] = s; __syncthreads();
    for (int st = 128; st > 0; st >>= 1) { if (tid < st) red[tid] += red[tid + st]; __syncthreads(); }
    if (tid == 0) part[blockIdx.x] = red[0];
}

__global__ void __launch_bounds__(256)
trace_final_kernel(double* __restrict__ accSlot, const double* __restrict__ part)
{
    __shared__ double red[256];
    const int tid = threadIdx.x;
    red[tid] = part[tid] + part[tid + 256];
    __syncthreads();
    for (int st = 128; st > 0; st >>= 1) { if (tid < st) red[tid] += red[tid + st]; __syncthreads(); }
    if (tid == 0) accSlot[0] = red[0];
}

__global__ void finalize_kernel(float* __restrict__ out, const double* __restrict__ acc)
{
    if (threadIdx.x == 0) out[0] = (float)(acc[0] - acc[1]);
}

// ---------------------------------------------------------------------------
// Host orchestration
// ---------------------------------------------------------------------------
extern "C" void kernel_launch(void* const* d_in, const int* in_sizes, int n_in,
                              void* d_out, int out_size, void* d_ws, size_t ws_size,
                              hipStream_t stream)
{
    (void)in_sizes; (void)n_in; (void)out_size; (void)ws_size;

    const float*  theta_p = (const float*)d_in[0];
    const float*  thA     = (const float*)d_in[1];
    const float*  thB     = (const float*)d_in[2];
    const double* rt      = (const double*)d_in[3];   // complex128 interleaved
    float* out = (float*)d_out;

    char* ws = (char*)d_ws;
    float*     p    = (float*)ws;                       // 2048 f32
    double*    part = (double*)(ws + 8 * 1024);         // 512 f64
    double*    acc  = (double*)(ws + 16 * 1024);        // 2 f64
    _Float16*  PsiR = (_Float16*)(ws + 64 * 1024);      // 2048x1024 f16
    _Float16*  PsiI = PsiR + (size_t)KSTATES * DD;
    float*     pl0  = (float*)(ws + 64 * 1024 + (size_t)2 * KSTATES * DD * sizeof(_Float16));
    const size_t PS = (size_t)DD * DD;
    // 14 f32 planes: RT(2) RHO(2) Y(2) Z(2) T(2) W(2) LOG(2)
    float *RTre = pl0 + 0*PS, *RTim = pl0 + 1*PS;
    float *RHre = pl0 + 2*PS, *RHim = pl0 + 3*PS;
    float *Y0r  = pl0 + 4*PS, *Y0i  = pl0 + 5*PS;
    float *Z0r  = pl0 + 6*PS, *Z0i  = pl0 + 7*PS;
    float *T0r  = pl0 + 8*PS, *T0i  = pl0 + 9*PS;
    float *Wr   = pl0 + 10*PS, *Wi  = pl0 + 11*PS;
    float *Lr   = pl0 + 12*PS, *Li  = pl0 + 13*PS;

    const int  EW_BLOCKS = (DD * DD) / 256;   // 4096
    const dim3 gemmGrid(DD / BM, DD / BN);    // 8 x 32

    auto cgemm = [&](float* cr, float* ci, const float* arr, const float* aii,
                     const float* brr, const float* bii) {
        cgemm_nn_kernel<<<gemmGrid, 256, 0, stream>>>(cr, ci, arr, aii, brr, bii);
    };
    auto affine = [&](float* wr, float* wi, const float* tr, const float* ti, float a, float b) {
        affine_diag_kernel<<<EW_BLOCKS, 256, 0, stream>>>(wr, wi, tr, ti, a, b);
    };
    auto axpy = [&](float* yr, float* yi, const float* xr, const float* xi, float a, float b) {
        axpy_kernel<<<EW_BLOCKS, 256, 0, stream>>>(yr, yi, xr, xi, a, b);
    };

    // ---- inputs -> p, Psi, rho, rho_target(f32) ----
    softmax_kernel  <<<1,       256, 0, stream>>>(p, theta_p);
    build_psi_kernel<<<KSTATES, 128, 0, stream>>>(PsiR, PsiI, thA, thB, p);
    rho_gemm_kernel <<<gemmGrid, 256, 0, stream>>>(RHre, RHim, PsiR, PsiI);
    conv_rt_kernel  <<<EW_BLOCKS, 256, 0, stream>>>(RTre, RTim, rt);

    // Gauss-Legendre order 8 on [0,1]
    const double gt[8] = {-0.9602898564975362, -0.7966664774136267, -0.5255324099163290,
                          -0.1834346424956498,  0.1834346424956498,  0.5255324099163290,
                           0.7966664774136267,  0.9602898564975362};
    const double gw[8] = { 0.1012285362903763,  0.2223810344533745,  0.3137066458778873,
                           0.3626837833783620,  0.3626837833783620,  0.3137066458778873,
                           0.2223810344533745,  0.1012285362903763};

    const int NS_ITERS = 12, INV_ITERS = 7, NUM_SQRTM = 6;

    // logm(A) via 6x Newton-Schulz sqrt + Pade partial fractions; then
    // acc[slot] = Re tr(rho_t @ 64*logB).  A planes are reused as E scratch.
    auto logm_trace = [&](float* Ar, float* Ai, double* accSlot) {
        float *yR = Y0r, *yI = Y0i, *zR = Z0r, *zI = Z0i, *tR = T0r, *tI = T0i;
        axpy(yR, yI, Ar, Ai, 1.0f, 0.0f);                     // Y = A
        for (int lvl = 0; lvl < NUM_SQRTM; ++lvl) {
            ident_kernel<<<EW_BLOCKS, 256, 0, stream>>>(zR, zI);   // Z = I
            for (int it = 0; it < NS_ITERS; ++it) {
                cgemm(tR, tI, zR, zI, yR, yI);                // T = Z*Y
                affine(Wr, Wi, tR, tI, -0.5f, 1.5f);          // W = 1.5I - 0.5T
                cgemm(tR, tI, yR, yI, Wr, Wi);                // Y' = Y*W  (into T pair)
                cgemm(yR, yI, Wr, Wi, zR, zI);                // Z' = W*Z  (into old Y pair)
                float* sw;
                sw = tR; tR = zR; zR = yR; yR = sw;           // rotate: Y<-T, Z<-oldY, T<-oldZ
                sw = tI; tI = zI; zI = yI; yI = sw;
            }
            // Y now holds the next-level sqrt
        }
        affine(Ar, Ai, yR, yI, 1.0f, -1.0f);                  // E = B - I  (into A planes)
        float *eR = Ar, *eI = Ai;
        axpy(Lr, Li, Lr, Li, 0.0f, 0.0f);                     // LOG = 0
        for (int jn = 0; jn < 8; ++jn) {
            const float xj = (float)(0.5 * (gt[jn] + 1.0));
            const float wj = (float)(0.5 * gw[jn] * 64.0);    // fold 2^NUM_SQRTM
            affine(zR, zI, eR, eI, xj, 1.0f);                 // M  = I + xj*E  (Z pair)
            affine(yR, yI, zR, zI, -1.0f, 2.0f);              // X0 = 2I - M    (Y pair)
            float *xR = yR, *xI = yI, *sR = tR, *sI = tI;
            for (int it = 0; it < INV_ITERS; ++it) {
                cgemm(sR, sI, zR, zI, xR, xI);                // T1 = M*X
                affine(Wr, Wi, sR, sI, -1.0f, 2.0f);          // W  = 2I - T1
                cgemm(sR, sI, xR, xI, Wr, Wi);                // X' = X*W
                float* sw;
                sw = xR; xR = sR; sR = sw;
                sw = xI; xI = sI; sI = sw;
            }
            cgemm(sR, sI, xR, xI, eR, eI);                    // S = X*E
            axpy(Lr, Li, sR, sI, wj, 1.0f);                   // LOG += 64*w_j*S
        }
        trace_partial_kernel<<<512, 256, 0, stream>>>(part, rt, Lr, Li);
        trace_final_kernel  <<<1,   256, 0, stream>>>(accSlot, part);
    };

    logm_trace(RTre, RTim, acc + 0);   // t1 = tr(rho_t log rho_t)
    logm_trace(RHre, RHim, acc + 1);   // t2 = Re tr(rho_t log rho)
    finalize_kernel<<<1, 32, 0, stream>>>(out, acc);          // loss = t1 - t2
}
